// RSSMRollout_52201032516219
// MI455X (gfx1250) — matrix operand order repaired and checked by reading, not compiled, not run on hardware
//
#include <hip/hip_runtime.h>

#define T_STEPS 50
#define B_SZ    2048
#define A_SZ    6
#define S_SZ    30
#define D_SZ    200
#define H_SZ    200
#define E_SZ    1024

#define HP   224   // padded H/D (7*32)
#define ASP  64    // padded A+S
#define QKP  1248  // padded D+E (224 + 1024)
#define NP2  64    // padded 2S
#define ROWS 16
#define FDIM 290   // 3S + D

typedef __attribute__((ext_vector_type(16))) __bf16        v16bf;
typedef __attribute__((ext_vector_type(8)))  float         v8f;
typedef __attribute__((ext_vector_type(4)))  unsigned int  u32x4;
typedef __attribute__((ext_vector_type(4)))  float         f32x4;

union FragU { v16bf v; u32x4 u[2]; };

__device__ __forceinline__ v8f wmma_bf16(v16bf a, v16bf b, v8f c) {
  return __builtin_amdgcn_wmma_f32_16x16x32_bf16(false, a, false, b, (short)0, c, false, false);
}

// A (16xK) fragment, row-major [16][ldk] bf16 in LDS.
// Per ISA 16-bit A layout: lane m=L&15, half=L>>4; elems 0..7 -> K = k0+8*half+0..7,
// elems 8..15 -> K = k0+16+8*half+0..7  (two 16B loads).
__device__ __forceinline__ v16bf load_a(const __bf16* base, int ldk, int k0, int lane) {
  int m = lane & 15, half = lane >> 4;
  const __bf16* p = base + m * ldk + k0 + 8 * half;
  FragU f;
  f.u[0] = *(const u32x4*)(p);
  f.u[1] = *(const u32x4*)(p + 16);
  return f.v;
}

// B (32x16) fragment from packed weights WT[N][Kp] bf16 (column of original matrix = row here).
// lane n=L&15, half=L>>4; elems 0..7 -> K = kt*32 + 8*half + 0..7, elems 8..15 -> K = kt*32 + 16 + 8*half + 0..7 (two 16B loads).
__device__ __forceinline__ v16bf load_b(const __bf16* wt, int kp, int ntile, int kt, int lane) {
  int n = lane & 15, half = lane >> 4;
  const __bf16* p = wt + (size_t)(ntile * 16 + n) * kp + kt * 32 + 8 * half;
  FragU f;
  f.u[0] = *(const u32x4*)(p);
  f.u[1] = *(const u32x4*)(p + 16);
  return f.v;
}

__device__ __forceinline__ v8f bias_init(const float* b, int col) {
  float v = b[col];
  v8f c;
#pragma unroll
  for (int i = 0; i < 8; ++i) c[i] = v;
  return c;
}

// ---- branchless fast activations (single TRANS ops, co-execute with WMMA) ----
__device__ __forceinline__ float fexp(float x)  { return __expf(x); }      // v_exp_f32 path
__device__ __forceinline__ float frcp(float x)  { return __builtin_amdgcn_rcpf(x); }

__device__ __forceinline__ float eluf(float x) {
  float e = fexp(fminf(x, 0.f)) - 1.f;        // argument clamped: no overflow ever
  return x > 0.f ? x : e;                     // if-converted to v_cndmask
}
__device__ __forceinline__ float sigm(float x) {
  return frcp(1.f + fexp(-x));
}
__device__ __forceinline__ float tanh_fast(float x) {
  float xc = fminf(fmaxf(x, -15.f), 15.f);    // avoid inf/inf
  float e  = fexp(2.f * xc);
  return (e - 1.f) * frcp(e + 1.f);
}
__device__ __forceinline__ float softplusf(float x) {
  float e  = fexp(fminf(x, 20.f));
  float sp = __logf(1.f + e);                 // v_log_f32 path
  return x > 20.f ? x : sp;
}

__global__ __launch_bounds__(256) void rssm_rollout_kernel(
    const float* __restrict__ obs, const float* __restrict__ act,
    const float* __restrict__ prev_stoch, const float* __restrict__ prev_deter,
    const float* __restrict__ eps_p, const float* __restrict__ eps_q,
    const __bf16* __restrict__ W1T,  const __bf16* __restrict__ WihT,
    const __bf16* __restrict__ WhhT, const __bf16* __restrict__ Wp1T,
    const __bf16* __restrict__ Wp2T, const __bf16* __restrict__ Wq1T,
    const __bf16* __restrict__ Wq2T, const float* __restrict__ bias,
    float* __restrict__ out)
{
  __shared__ __align__(16) __bf16 s_xa[ROWS * ASP];
  __shared__ __align__(16) __bf16 s_det[2][ROWS * HP];
  __shared__ __align__(16) float  s_detf[ROWS * HP];
  __shared__ __align__(16) __bf16 s_act[ROWS * HP];
  __shared__ __align__(16) __bf16 s_obs[ROWS * 512];
  __shared__ __align__(16) float  s_out[ROWS * NP2];

  const int tid  = threadIdx.x;
  const int lane = tid & 31, wid = tid >> 5;
  const int l15  = lane & 15, half = lane >> 4;
  const int rb   = blockIdx.x * ROWS;

  const float* b1p  = bias + 0;
  const float* bihp = bias + 224;
  const float* bhhp = bias + 896;
  const float* bp1p = bias + 1568;
  const float* bp2p = bias + 1792;
  const float* bq1p = bias + 1856;
  const float* bq2p = bias + 2080;

  float* prior = out;
  float* post  = out + (size_t)T_STEPS * B_SZ * FDIM;

  // ---- init carry state ----
  for (int i = tid; i < ROWS * ASP; i += 256) s_xa[i] = (__bf16)0.f;
  for (int i = tid; i < ROWS * HP; i += 256) {
    int r = i / HP, j = i % HP;
    float d = (j < D_SZ) ? prev_deter[(size_t)(rb + r) * D_SZ + j] : 0.f;
    s_det[0][i] = (__bf16)d;
    s_detf[i]   = d;
  }
  __syncthreads();
  for (int i = tid; i < ROWS * S_SZ; i += 256) {
    int r = i / S_SZ, s = i % S_SZ;
    s_xa[r * ASP + A_SZ + s] = (__bf16)prev_stoch[(size_t)(rb + r) * S_SZ + s];
  }
  for (int i = tid; i < ROWS * A_SZ; i += 256) {
    int r = i / A_SZ, c = i % A_SZ;
    s_xa[r * ASP + c] = (__bf16)act[(size_t)(rb + r) * A_SZ + c];
  }
  __syncthreads();

  for (int t = 0; t < T_STEPS; ++t) {
    const int cur = t & 1, nxt = cur ^ 1;

    // prefetch next step's obs rows (256 threads x 256B = 64KB tile)
    if (t + 1 < T_STEPS) {
      const float* nf = obs + ((size_t)(t + 1) * B_SZ + rb) * E_SZ + (size_t)tid * 64;
      __builtin_prefetch(nf, 0, 1);
    }

    // ---- GEMM1: x = ELU(xa @ W1 + b1) -> s_act ----
    for (int nt = wid; nt < 14; nt += 8) {
      v8f acc = bias_init(b1p, nt * 16 + l15);
#pragma unroll
      for (int kt = 0; kt < 2; ++kt) {
        v16bf a = load_a(s_xa, ASP, kt * 32, lane);
        acc = wmma_bf16(a, load_b(W1T, ASP, nt, kt, lane), acc);
      }
#pragma unroll
      for (int v = 0; v < 8; ++v)
        s_act[(v + 8 * half) * HP + nt * 16 + l15] = (__bf16)eluf(acc[v]);
    }
    __syncthreads();

    // ---- GEMM2 (gi & gh) + GRU -> s_det[nxt], s_detf (in place) ----
    for (int nt = wid; nt < 14; nt += 8) {
      int col = nt * 16 + l15;
      v8f air = bias_init(bihp, col);
      v8f aiz = bias_init(bihp, 224 + col);
      v8f ain = bias_init(bihp, 448 + col);
      v8f ahr = bias_init(bhhp, col);
      v8f ahz = bias_init(bhhp, 224 + col);
      v8f ahn = bias_init(bhhp, 448 + col);
      for (int kt = 0; kt < 7; ++kt) {
        v16bf ax = load_a(s_act, HP, kt * 32, lane);
        v16bf ah = load_a(s_det[cur], HP, kt * 32, lane);
        air = wmma_bf16(ax, load_b(WihT, HP, nt,      kt, lane), air);
        aiz = wmma_bf16(ax, load_b(WihT, HP, 14 + nt, kt, lane), aiz);
        ain = wmma_bf16(ax, load_b(WihT, HP, 28 + nt, kt, lane), ain);
        ahr = wmma_bf16(ah, load_b(WhhT, HP, nt,      kt, lane), ahr);
        ahz = wmma_bf16(ah, load_b(WhhT, HP, 14 + nt, kt, lane), ahz);
        ahn = wmma_bf16(ah, load_b(WhhT, HP, 28 + nt, kt, lane), ahn);
      }
#pragma unroll
      for (int v = 0; v < 8; ++v) {
        int m   = v + 8 * half;
        float r = sigm(air[v] + ahr[v]);
        float z = sigm(aiz[v] + ahz[v]);
        float n = tanh_fast(ain[v] + r * ahn[v]);
        float d = s_detf[m * HP + col];
        float h = (1.f - z) * n + z * d;
        s_detf[m * HP + col]       = h;           // sole owner of (m,col): safe in place
        s_det[nxt][m * HP + col]   = (__bf16)h;
      }
    }
    __syncthreads();

    // ---- GEMM3: ph = ELU(h @ Wp1 + bp1) -> s_act ----
    for (int nt = wid; nt < 14; nt += 8) {
      v8f acc = bias_init(bp1p, nt * 16 + l15);
      for (int kt = 0; kt < 7; ++kt) {
        v16bf a = load_a(s_det[nxt], HP, kt * 32, lane);
        acc = wmma_bf16(a, load_b(Wp1T, HP, nt, kt, lane), acc);
      }
#pragma unroll
      for (int v = 0; v < 8; ++v)
        s_act[(v + 8 * half) * HP + nt * 16 + l15] = (__bf16)eluf(acc[v]);
    }
    __syncthreads();

    // ---- GEMM4: pout = ph @ Wp2 + bp2 -> s_out ----
    if (wid < 4) {
      v8f acc = bias_init(bp2p, wid * 16 + l15);
      for (int kt = 0; kt < 7; ++kt) {
        v16bf a = load_a(s_act, HP, kt * 32, lane);
        acc = wmma_bf16(a, load_b(Wp2T, HP, wid, kt, lane), acc);
      }
#pragma unroll
      for (int v = 0; v < 8; ++v)
        s_out[(v + 8 * half) * NP2 + wid * 16 + l15] = acc[v];
    }
    __syncthreads();

    // ---- prior elementwise + deterministic outputs ----
    for (int i = tid; i < ROWS * S_SZ; i += 256) {
      int r = i / S_SZ, s = i % S_SZ;
      size_t b = (size_t)rb + r;
      float pm = s_out[r * NP2 + s];
      float ps = softplusf(s_out[r * NP2 + S_SZ + s]) + 0.1f;
      float ep = eps_p[((size_t)t * B_SZ + b) * S_SZ + s];
      float* o = prior + ((size_t)t * B_SZ + b) * FDIM;
      o[s] = pm; o[S_SZ + s] = ps; o[2 * S_SZ + s] = pm + ps * ep;
    }
    for (int i = tid; i < ROWS * D_SZ; i += 256) {
      int r = i / D_SZ, j = i % D_SZ;
      size_t b = (size_t)rb + r;
      float h = s_detf[r * HP + j];
      prior[((size_t)t * B_SZ + b) * FDIM + 3 * S_SZ + j] = h;
      post [((size_t)t * B_SZ + b) * FDIM + 3 * S_SZ + j] = h;
    }

    // ---- GEMM5: qh = ELU([h|obs] @ Wq1 + bq1) -> s_act ----
    const bool has1 = (wid + 8) < 14;
    v8f q0 = bias_init(bq1p, wid * 16 + l15);
    v8f q1 = bias_init(bq1p, has1 ? ((wid + 8) * 16 + l15) : l15);
    for (int kt = 0; kt < 7; ++kt) {  // K chunk 0: deter part (cols 0..223)
      v16bf a = load_a(s_det[nxt], HP, kt * 32, lane);
      q0 = wmma_bf16(a, load_b(Wq1T, QKP, wid, kt, lane), q0);
      if (has1) q1 = wmma_bf16(a, load_b(Wq1T, QKP, wid + 8, kt, lane), q1);
    }
    for (int ch = 0; ch < 2; ++ch) {  // obs part staged in 2 x 512 col chunks
      __syncthreads();
      const float* src = obs + ((size_t)t * B_SZ + rb) * E_SZ + ch * 512;
      for (int i = tid; i < ROWS * 128; i += 256) {
        int r = i >> 7, c4 = i & 127;
        f32x4 f = *(const f32x4*)(src + (size_t)r * E_SZ + c4 * 4);
        __bf16* d = s_obs + r * 512 + c4 * 4;
        d[0] = (__bf16)f.x; d[1] = (__bf16)f.y; d[2] = (__bf16)f.z; d[3] = (__bf16)f.w;
      }
      __syncthreads();
      for (int kt = 0; kt < 16; ++kt) {
        int ktg = 7 + ch * 16 + kt;
        v16bf a = load_a(s_obs, 512, kt * 32, lane);
        q0 = wmma_bf16(a, load_b(Wq1T, QKP, wid, ktg, lane), q0);
        if (has1) q1 = wmma_bf16(a, load_b(Wq1T, QKP, wid + 8, ktg, lane), q1);
      }
    }
    __syncthreads();
#pragma unroll
    for (int v = 0; v < 8; ++v) {
      int m = v + 8 * half;
      s_act[m * HP + wid * 16 + l15] = (__bf16)eluf(q0[v]);
      if (has1) s_act[m * HP + (wid + 8) * 16 + l15] = (__bf16)eluf(q1[v]);
    }
    __syncthreads();

    // ---- GEMM6: qout = qh @ Wq2 + bq2 -> s_out ----
    if (wid < 4) {
      v8f acc = bias_init(bq2p, wid * 16 + l15);
      for (int kt = 0; kt < 7; ++kt) {
        v16bf a = load_a(s_act, HP, kt * 32, lane);
        acc = wmma_bf16(a, load_b(Wq2T, HP, wid, kt, lane), acc);
      }
#pragma unroll
      for (int v = 0; v < 8; ++v)
        s_out[(v + 8 * half) * NP2 + wid * 16 + l15] = acc[v];
    }
    __syncthreads();

    // ---- posterior elementwise: outputs + next stoch carry ----
    for (int i = tid; i < ROWS * S_SZ; i += 256) {
      int r = i / S_SZ, s = i % S_SZ;
      size_t b = (size_t)rb + r;
      float qm  = s_out[r * NP2 + s];
      float qs  = softplusf(s_out[r * NP2 + S_SZ + s]) + 0.1f;
      float eq  = eps_q[((size_t)t * B_SZ + b) * S_SZ + s];
      float qst = qm + qs * eq;
      float* o  = post + ((size_t)t * B_SZ + b) * FDIM;
      o[s] = qm; o[S_SZ + s] = qs; o[2 * S_SZ + s] = qst;
      s_xa[r * ASP + A_SZ + s] = (__bf16)qst;
    }
    if (t + 1 < T_STEPS) {
      for (int i = tid; i < ROWS * A_SZ; i += 256) {
        int r = i / A_SZ, c = i % A_SZ;
        s_xa[r * ASP + c] = (__bf16)act[((size_t)(t + 1) * B_SZ + rb + r) * A_SZ + c];
      }
    }
    __syncthreads();
  }
}

// ---- weight packing: zero-padded, transposed-to-[N][Kp] bf16 ----
__global__ void pack_w_kernel(__bf16* dst, const float* src, int dstN, int dstK, int mode) {
  int i = blockIdx.x * 256 + threadIdx.x;
  if (i >= dstN * dstK) return;
  int n = i / dstK, k = i % dstK;
  float v = 0.f;
  switch (mode) {
    case 0:  // W1 [36][200] K-major
      if (n < 200 && k < 36) v = src[k * 200 + n];
      break;
    case 1: {  // w_ih / w_hh [600][200] N-major, 3 gates each padded 200->224
      int g = n / 224, c = n % 224;
      if (c < 200 && k < 200) v = src[(g * 200 + c) * 200 + k];
    } break;
    case 2:  // Wp1 [200][200] K-major
      if (n < 200 && k < 200) v = src[k * 200 + n];
      break;
    case 3:  // Wp2 / Wq2 [200][60] K-major
      if (n < 60 && k < 200) v = src[k * 60 + n];
      break;
    case 4: {  // Wq1 [1224][200] K-major; K remap: 0..199 -> h, 200..223 pad, 224.. -> obs
      int ks = (k < 200) ? k : ((k >= 224) ? (k - 24) : -1);
      if (n < 200 && ks >= 0 && ks < 1224) v = src[ks * 200 + n];
    } break;
  }
  dst[i] = (__bf16)v;
}

__global__ void pack_b_kernel(float* dst, const float* src, int dstN, int srcUnit, int pad) {
  int i = blockIdx.x * 256 + threadIdx.x;
  if (i >= dstN) return;
  int g = i / pad, c = i % pad;
  dst[i] = (c < srcUnit) ? src[g * srcUnit + c] : 0.f;
}

extern "C" void kernel_launch(void* const* d_in, const int* in_sizes, int n_in,
                              void* d_out, int out_size, void* d_ws, size_t ws_size,
                              hipStream_t stream) {
  (void)in_sizes; (void)n_in; (void)out_size; (void)ws_size;
  const float* obs    = (const float*)d_in[1];
  const float* act    = (const float*)d_in[2];
  const float* pstoch = (const float*)d_in[3];
  const float* pdet   = (const float*)d_in[4];
  const float* epsp   = (const float*)d_in[5];
  const float* epsq   = (const float*)d_in[6];
  const float* W1  = (const float*)d_in[7];
  const float* b1  = (const float*)d_in[8];
  const float* wih = (const float*)d_in[9];
  const float* bih = (const float*)d_in[10];
  const float* whh = (const float*)d_in[11];
  const float* bhh = (const float*)d_in[12];
  const float* Wp1 = (const float*)d_in[13];
  const float* bp1 = (const float*)d_in[14];
  const float* Wp2 = (const float*)d_in[15];
  const float* bp2 = (const float*)d_in[16];
  const float* Wq1 = (const float*)d_in[17];
  const float* bq1 = (const float*)d_in[18];
  const float* Wq2 = (const float*)d_in[19];
  const float* bq2 = (const float*)d_in[20];

  char* ws = (char*)d_ws;
  size_t off = 0;
  auto alloc_bf = [&](size_t elems) { __bf16* p = (__bf16*)(ws + off); off += elems * 2; return p; };
  __bf16* W1T  = alloc_bf(224 * 64);
  __bf16* WihT = alloc_bf(672 * 224);
  __bf16* WhhT = alloc_bf(672 * 224);
  __bf16* Wp1T = alloc_bf(224 * 224);
  __bf16* Wp2T = alloc_bf(64 * 224);
  __bf16* Wq1T = alloc_bf(224 * 1248);
  __bf16* Wq2T = alloc_bf(64 * 224);
  off = (off + 255) & ~(size_t)255;
  float* bias = (float*)(ws + off);

  auto nb = [](int n) { return dim3((unsigned)((n + 255) / 256)); };
  pack_w_kernel<<<nb(224 * 64),   256, 0, stream>>>(W1T,  W1,  224, 64,   0);
  pack_w_kernel<<<nb(672 * 224),  256, 0, stream>>>(WihT, wih, 672, 224,  1);
  pack_w_kernel<<<nb(672 * 224),  256, 0, stream>>>(WhhT, whh, 672, 224,  1);
  pack_w_kernel<<<nb(224 * 224),  256, 0, stream>>>(Wp1T, Wp1, 224, 224,  2);
  pack_w_kernel<<<nb(64 * 224),   256, 0, stream>>>(Wp2T, Wp2, 64,  224,  3);
  pack_w_kernel<<<nb(224 * 1248), 256, 0, stream>>>(Wq1T, Wq1, 224, 1248, 4);
  pack_w_kernel<<<nb(64 * 224),   256, 0, stream>>>(Wq2T, Wq2, 64,  224,  3);

  pack_b_kernel<<<nb(224), 256, 0, stream>>>(bias + 0,    b1,  224, 200, 224);
  pack_b_kernel<<<nb(672), 256, 0, stream>>>(bias + 224,  bih, 672, 200, 224);
  pack_b_kernel<<<nb(672), 256, 0, stream>>>(bias + 896,  bhh, 672, 200, 224);
  pack_b_kernel<<<nb(224), 256, 0, stream>>>(bias + 1568, bp1, 224, 200, 224);
  pack_b_kernel<<<nb(64),  256, 0, stream>>>(bias + 1792, bp2, 64,  60,  64);
  pack_b_kernel<<<nb(224), 256, 0, stream>>>(bias + 1856, bq1, 224, 200, 224);
  pack_b_kernel<<<nb(64),  256, 0, stream>>>(bias + 2080, bq2, 64,  60,  64);

  rssm_rollout_kernel<<<dim3(B_SZ / ROWS), 256, 0, stream>>>(
      obs, act, pstoch, pdet, epsp, epsq,
      W1T, WihT, WhhT, Wp1T, Wp2T, Wq1T, Wq2T, bias, (float*)d_out);
}
